// MutiHeadAttention_63299228009350
// MI455X (gfx1250) — matrix operand order repaired and checked by reading, not compiled, not run on hardware
//
#include <hip/hip_runtime.h>

#define DIMM  512
#define HEADS 8
#define DHEAD 64
#define BATCH 2
#define SEQ   2048
#define MTOT  (BATCH*SEQ)   // 4096
#define KVT   64            // kv tile width in attention

typedef __attribute__((ext_vector_type(16))) __bf16 v16bf;
typedef __attribute__((ext_vector_type(8)))  float  v8f;

// ---------------------------------------------------------------------------
// Fragment loaders (CDNA5 wave32 WMMA 16x16x32 bf16 VGPR layouts, ISA 7.12.2)
// A-matrix 16x32: lane m=lane&15, half=lane>>4; VGPR i holds K pair
//   k = (i<4 ? 2i : 16+2(i-4)) + 8*half
// B-matrix 32x16: lane n=lane&15, half=lane>>4; VGPR i holds K pair
//   k = (i<4 ? 2i : 16+2(i-4)) + 8*half   (same K order as A)
// ---------------------------------------------------------------------------
__device__ inline v16bf load_fragA(const __bf16* base, int stride, int lm, int half, int kbase) {
  v16bf f;
#pragma unroll
  for (int i = 0; i < 8; ++i) {
    int k = kbase + ((i < 4) ? (2 * i) : (16 + 2 * (i - 4))) + 8 * half;
    const __bf16* p = base + lm * stride + k;
    f[2 * i]     = p[0];
    f[2 * i + 1] = p[1];
  }
  return f;
}

__device__ inline v16bf load_fragB(const __bf16* base, int stride, int lm, int half, int kbase) {
  v16bf f;
#pragma unroll
  for (int i = 0; i < 8; ++i) {
    int k = kbase + ((i < 4) ? (2 * i) : (16 + 2 * (i - 4))) + 8 * half;
    const __bf16* p = base + lm * stride + k;   // row = N column index lm
    f[2 * i]     = p[0];
    f[2 * i + 1] = p[1];
  }
  return f;
}

// ---------------------------------------------------------------------------
// GEMM: C[m,n] = sum_k A[m,k] * W[n,k] + bias[n]   (torch Linear, W row-major [N][K])
// MODE 0: A = f32, out -> bf16 head-split [(b*H+h)*S + s][d]
// MODE 1: A = bf16 (ctx), out -> f32 plain [m][n]
// grid (MTOT/64, DIMM/64), 128 threads (4 waves); wave owns 16x64 strip.
// ---------------------------------------------------------------------------
template <int MODE>
__global__ __launch_bounds__(128) void gemm_wmma_kernel(
    const float* __restrict__ Af32, const __bf16* __restrict__ Abf,
    const float* __restrict__ W, const float* __restrict__ bias,
    __bf16* __restrict__ out_bf, float* __restrict__ out_f32)
{
  const int m0   = blockIdx.x * 64;
  const int n0   = blockIdx.y * 64;
  const int tid  = threadIdx.x;
  const int wave = tid >> 5;
  const int lane = tid & 31;
  const int lm   = lane & 15;
  const int half = lane >> 4;

  __shared__ __bf16 At[64 * 34];   // [row][k], stride 34 (pad vs bank conflicts)
  __shared__ __bf16 Wt[64 * 34];   // [n][k]

  v8f acc[4] = {};

  for (int k0 = 0; k0 < DIMM; k0 += 32) {
    for (int idx = tid; idx < 64 * 32; idx += 128) {
      int r = idx >> 5, c = idx & 31;
      float av = (MODE == 0) ? Af32[(size_t)(m0 + r) * DIMM + k0 + c]
                             : (float)Abf[(size_t)(m0 + r) * DIMM + k0 + c];
      At[r * 34 + c] = (__bf16)av;
      Wt[r * 34 + c] = (__bf16)W[(size_t)(n0 + r) * DIMM + k0 + c];
    }
    if (k0 + 32 < DIMM) {   // prefetch next W tile -> global_prefetch_b8
      __builtin_prefetch(&W[(size_t)(n0 + (tid >> 1)) * DIMM + k0 + 32 + (tid & 1) * 16], 0, 1);
    }
    __syncthreads();

    v16bf a = load_fragA(&At[(wave * 16) * 34], 34, lm, half, 0);
#pragma unroll
    for (int f = 0; f < 4; ++f) {
      v16bf b = load_fragB(&Wt[(f * 16) * 34], 34, lm, half, 0);
      acc[f] = __builtin_amdgcn_wmma_f32_16x16x32_bf16(false, a, false, b,
                                                       (short)0, acc[f], false, false);
    }
    __syncthreads();
  }

  const int mrow = m0 + wave * 16;
#pragma unroll
  for (int f = 0; f < 4; ++f) {
#pragma unroll
    for (int j = 0; j < 8; ++j) {
      int M  = j + 8 * half;          // C layout: VGPR j, lanes16-31 hold M+8
      int gm = mrow + M;
      int gn = n0 + f * 16 + lm;
      float v = acc[f][j] + bias[gn];
      if (MODE == 0) {
        int bb = gm >> 11, s = gm & (SEQ - 1);
        int h  = gn >> 6,  d = gn & (DHEAD - 1);
        out_bf[(((size_t)(bb * HEADS + h) * SEQ + s) << 6) + d] = (__bf16)v;
      } else {
        out_f32[(size_t)gm * DIMM + gn] = v;
      }
    }
  }
}

// ---------------------------------------------------------------------------
// Flash attention per (b,h): one WG per (bh, 64-query-row tile), 128 thr.
// KVT=64 kv rows per iteration, staged in LDS and shared by all 4 waves.
// K tile staged with GLOBAL_LOAD_ASYNC_TO_LDS_B128 (ASYNCcnt path);
// V tile transposed manually during staging.
// ---------------------------------------------------------------------------
__global__ __launch_bounds__(128) void attn_wmma_kernel(
    const __bf16* __restrict__ Qb, const __bf16* __restrict__ Kb,
    const __bf16* __restrict__ Vb, __bf16* __restrict__ Ctx)
{
  const int bh   = blockIdx.y;        // 0..15
  const int qt   = blockIdx.x;        // 0..31
  const int tid  = threadIdx.x;
  const int wave = tid >> 5;
  const int lane = tid & 31;
  const int lm   = lane & 15;
  const int half = lane >> 4;
  const int q0   = qt * 64 + wave * 16;

  const __bf16* Qh = Qb + (size_t)bh * SEQ * DHEAD;
  const __bf16* Kh = Kb + (size_t)bh * SEQ * DHEAD;
  const __bf16* Vh = Vb + (size_t)bh * SEQ * DHEAD;

  __shared__ __bf16 Kt[KVT * 66];       // [kv row][d], stride 66
  __shared__ __bf16 Vt[DHEAD * 66];     // [d][kv row], stride 66 (transposed)
  __shared__ __bf16 Pt[4][16 * 66];     // per-wave P scratch, [row][ks], stride 66

  // Q fragments (16 rows x 64 d), d halves 0..31 / 32..63, direct from global
  v16bf aq0 = load_fragA(Qh + (size_t)q0 * DHEAD, DHEAD, lm, half, 0);
  v16bf aq1 = load_fragA(Qh + (size_t)q0 * DHEAD, DHEAD, lm, half, 32);

  v8f   oacc[4] = {};
  float mrun[8], lrun[8];
#pragma unroll
  for (int j = 0; j < 8; ++j) { mrun[j] = -3.0e38f; lrun[j] = 0.0f; }

  const float inv_sqrt_d = 0.125f;     // 1/sqrt(64)

  for (int t = 0; t < SEQ / KVT; ++t) {
    const int kv0 = t * KVT;

    // --- async stage of K tile: 64x64 bf16 = 512 x 16B chunks, 4 per thread
#pragma unroll
    for (int it = 0; it < 4; ++it) {
      int idx = tid + it * 128;                 // 0..511
      int r   = idx >> 3;                       // kv row 0..63
      int c8  = (idx & 7) * 8;                  // col chunk (8 bf16 = 16B)
      unsigned lds_off = (unsigned)(uintptr_t)(&Kt[r * 66 + c8]);
      unsigned voff    = (unsigned)(((kv0 + r) * DHEAD + c8) * sizeof(__bf16));
      asm volatile("global_load_async_to_lds_b128 %0, %1, %2"
                   :: "v"(lds_off), "v"(voff), "s"(Kh) : "memory");
    }

    // --- stage V transposed (regular path; transpose can't be DMA'd)
    for (int idx = tid; idx < KVT * DHEAD; idx += 128) {
      int r = idx >> 6, c = idx & 63;
      Vt[c * 66 + r] = Vh[(size_t)(kv0 + r) * DHEAD + c];
    }
    if (t + 1 < SEQ / KVT) {                    // prefetch next V tile
      __builtin_prefetch(&Vh[(size_t)(kv0 + KVT + (tid >> 1)) * DHEAD + (tid & 1) * 32], 0, 1);
    }
    asm volatile("s_wait_asynccnt 0x0" ::: "memory");
    __syncthreads();

    // --- scores 16x64: S[q,ks] = sum_d Q[q,d] K[ks,d]; B[k=d][n=ks] = Kt[ks][d]
    v8f sc[4] = {};
#pragma unroll
    for (int nb = 0; nb < 4; ++nb) {
      v16bf b0 = load_fragB(&Kt[(nb * 16) * 66], 66, lm, half, 0);
      sc[nb] = __builtin_amdgcn_wmma_f32_16x16x32_bf16(false, aq0, false, b0,
                                                       (short)0, sc[nb], false, false);
      v16bf b1 = load_fragB(&Kt[(nb * 16) * 66], 66, lm, half, 32);
      sc[nb] = __builtin_amdgcn_wmma_f32_16x16x32_bf16(false, aq1, false, b1,
                                                       (short)0, sc[nb], false, false);
    }

    // --- online softmax: rows live in VGPR j across the 16 lanes of each half
#pragma unroll
    for (int j = 0; j < 8; ++j) {
      float s0 = sc[0][j] * inv_sqrt_d;
      float s1 = sc[1][j] * inv_sqrt_d;
      float s2 = sc[2][j] * inv_sqrt_d;
      float s3 = sc[3][j] * inv_sqrt_d;
      float rv = fmaxf(fmaxf(s0, s1), fmaxf(s2, s3));
#pragma unroll
      for (int off = 1; off < 16; off <<= 1) rv = fmaxf(rv, __shfl_xor(rv, off, 32));
      float mnew  = fmaxf(mrun[j], rv);
      float alpha = __expf(mrun[j] - mnew);
      float p0 = __expf(s0 - mnew);
      float p1 = __expf(s1 - mnew);
      float p2 = __expf(s2 - mnew);
      float p3 = __expf(s3 - mnew);
      float rs = (p0 + p1) + (p2 + p3);
#pragma unroll
      for (int off = 1; off < 16; off <<= 1) rs += __shfl_xor(rs, off, 32);
      lrun[j] = lrun[j] * alpha + rs;
      mrun[j] = mnew;
#pragma unroll
      for (int f = 0; f < 4; ++f) oacc[f][j] = oacc[f][j] * alpha;
      int M = j + 8 * half;
      Pt[wave][M * 66 +  0 + lm] = (__bf16)p0;
      Pt[wave][M * 66 + 16 + lm] = (__bf16)p1;
      Pt[wave][M * 66 + 32 + lm] = (__bf16)p2;
      Pt[wave][M * 66 + 48 + lm] = (__bf16)p3;
    }
    __syncthreads();   // Pt visibility + keep waves in step

    // --- ctx += P(16x64) * V(64x64); B[k=ks][n=d] = Vt[d][ks]
    v16bf ap0 = load_fragA(&Pt[wave][0], 66, lm, half, 0);
    v16bf ap1 = load_fragA(&Pt[wave][0], 66, lm, half, 32);
#pragma unroll
    for (int f = 0; f < 4; ++f) {
      v16bf bv0 = load_fragB(&Vt[(f * 16) * 66], 66, lm, half, 0);
      oacc[f] = __builtin_amdgcn_wmma_f32_16x16x32_bf16(false, ap0, false, bv0,
                                                        (short)0, oacc[f], false, false);
      v16bf bv1 = load_fragB(&Vt[(f * 16) * 66], 66, lm, half, 32);
      oacc[f] = __builtin_amdgcn_wmma_f32_16x16x32_bf16(false, ap1, false, bv1,
                                                        (short)0, oacc[f], false, false);
    }
    __syncthreads();   // protect Kt/Vt before next stage
  }

  // write ctx, merged layout [b][s][h*64+d]
  const int bb = bh >> 3, h = bh & 7;
#pragma unroll
  for (int f = 0; f < 4; ++f) {
#pragma unroll
    for (int j = 0; j < 8; ++j) {
      int M   = j + 8 * half;
      int s   = q0 + M;
      int col = h * DHEAD + f * 16 + lm;
      float v = oacc[f][j] / lrun[j];
      Ctx[(size_t)(bb * SEQ + s) * DIMM + col] = (__bf16)v;
    }
  }
}

// ---------------------------------------------------------------------------
extern "C" void kernel_launch(void* const* d_in, const int* in_sizes, int n_in,
                              void* d_out, int out_size, void* d_ws, size_t ws_size,
                              hipStream_t stream) {
  (void)in_sizes; (void)n_in; (void)out_size; (void)ws_size;

  const float* query = (const float*)d_in[0];
  const float* key   = (const float*)d_in[1];
  const float* value = (const float*)d_in[2];
  const float* Wq = (const float*)d_in[3];  const float* bq = (const float*)d_in[4];
  const float* Wk = (const float*)d_in[5];  const float* bk = (const float*)d_in[6];
  const float* Wv = (const float*)d_in[7];  const float* bv = (const float*)d_in[8];
  const float* Wo = (const float*)d_in[9];  const float* bo = (const float*)d_in[10];

  const size_t HEAD_ELEMS = (size_t)BATCH * HEADS * SEQ * DHEAD;  // 2M bf16 each
  __bf16* Qb  = (__bf16*)d_ws;
  __bf16* Kb  = Qb + HEAD_ELEMS;
  __bf16* Vb  = Kb + HEAD_ELEMS;
  __bf16* Ctx = Vb + HEAD_ELEMS;            // [4096][512] bf16; total ws = 16 MB

  dim3 ggrid(MTOT / 64, DIMM / 64);
  dim3 gblk(128);

  gemm_wmma_kernel<0><<<ggrid, gblk, 0, stream>>>(query, nullptr, Wq, bq, Qb, nullptr);
  gemm_wmma_kernel<0><<<ggrid, gblk, 0, stream>>>(key,   nullptr, Wk, bk, Kb, nullptr);
  gemm_wmma_kernel<0><<<ggrid, gblk, 0, stream>>>(value, nullptr, Wv, bv, Vb, nullptr);

  attn_wmma_kernel<<<dim3(SEQ / 64, BATCH * HEADS), gblk, 0, stream>>>(Qb, Kb, Vb, Ctx);

  gemm_wmma_kernel<1><<<ggrid, gblk, 0, stream>>>(nullptr, Ctx, Wo, bo, nullptr, (float*)d_out);
}
